// MultiHeadAttentionWithKVCache_27049704030457
// MI455X (gfx1250) — compile-verified, matrix operand order checked by reading, not run on hardware
//
#include <hip/hip_runtime.h>
#include <hip/hip_bf16.h>
#include <math.h>

typedef __attribute__((ext_vector_type(16))) __bf16 v16bf;
typedef __attribute__((ext_vector_type(8)))  __bf16 v8bf;
typedef __attribute__((ext_vector_type(4)))  __bf16 v4bf;
typedef __attribute__((ext_vector_type(8)))  float  v8f;
typedef __attribute__((ext_vector_type(4)))  unsigned int v4u;
typedef __attribute__((ext_vector_type(8)))  int    v8i;
typedef __attribute__((ext_vector_type(4)))  int    v4i;

#define D_MODEL 1024
#define NHEAD   16
#define HEAD_DIM 64
#define SEQ_B   2
#define SEQ_T   2048
#define MTOT    (SEQ_B * SEQ_T)   // 4096 rows

#if __has_builtin(__builtin_amdgcn_tensor_load_to_lds)
#  define HAVE_TDM 1
#  if __has_include(<hip/amd_detail/amd_gfx1250_TDM.h>)
#    define TDM_ARGS6 1      // therock-10.0 header toolchain -> 6-arg builtin
#  else
#    define TDM_ARGS6 0      // ROCm 7.2 clang-22 -> 5-arg builtin
#  endif
#else
#  define HAVE_TDM 0
#endif

// ---------------- helpers ----------------

// low 32 bits of a generic shared pointer are the LDS byte address (ISA 10.2)
__device__ __forceinline__ unsigned lds_off(const void* p) {
  return (unsigned)(unsigned long long)p;
}

#if HAVE_TDM
// 2D tile DMA: global (bf16, row stride stride_x elems) -> LDS with row padding.
// pad_ic: interval code (2^(c+1) dwords between pads); pad_ac: (pad dwords - 1).
__device__ __forceinline__ void tdm_load_2d(const void* gsrc, unsigned ldsaddr,
                                            unsigned tile_x, unsigned tile_y,
                                            unsigned stride_x,
                                            unsigned pad_ic, unsigned pad_ac) {
  unsigned long long ga = (unsigned long long)gsrc;
  v4u g0;
  g0[0] = 1u;                                        // count=1, user mode
  g0[1] = ldsaddr;                                   // lds_addr
  g0[2] = (unsigned)(ga & 0xffffffffu);              // global_addr[31:0]
  g0[3] = (unsigned)((ga >> 32) & 0x01ffffffu)       // global_addr[56:32]
        | (2u << 30);                                // type = 2 (image)
  v8i g1;
  g1[0] = (int)((1u << 16)                           // data_size = 2B
              | (1u << 20)                           // pad_enable
              | (pad_ic << 22) | (pad_ac << 25));
  g1[1] = (int)((tile_x & 0xffffu) << 16);           // tensor_dim0 = tile_x
  g1[2] = (int)(((tile_x >> 16) & 0xffffu) | ((tile_y & 0xffffu) << 16)); // tensor_dim1
  g1[3] = (int)(((tile_y >> 16) & 0xffffu) | ((tile_x & 0xffffu) << 16)); // tile_dim0
  g1[4] = (int)(tile_y & 0xffffu);                   // tile_dim1 (tile_dim2 = 0)
  g1[5] = (int)stride_x;                             // tensor_dim0_stride[31:0]
  g1[6] = 0;                                         // stride0 hi / stride1 lo
  g1[7] = 0;
  v4i z4 = {0, 0, 0, 0};
#if TDM_ARGS6
  v8i z8 = {0, 0, 0, 0, 0, 0, 0, 0};
  __builtin_amdgcn_tensor_load_to_lds(g0, g1, z4, z4, z8, 0);
#else
  __builtin_amdgcn_tensor_load_to_lds(g0, g1, z4, z4, 0);
#endif
}
#endif

// 16x32 bf16 A-style fragment from a row-major LDS tile (ISA 7.12.2).
__device__ __forceinline__ v16bf load_frag16(const __bf16* rowbase, int ldw, int lane) {
  const int h = (lane >> 4) & 1;
  const int m = lane & 15;
  const __bf16* p = rowbase + m * ldw + 8 * h;
  union { v16bf v; v8bf q[2]; } u;
  u.q[0] = *(const v8bf*)(p);
  u.q[1] = *(const v8bf*)(p + 16);
  return u.v;
}

// packed f32x8 -> bf16x8 (lowers to v_cvt_pk_bf16_f32)
__device__ __forceinline__ v8bf cvt8(v8f v) {
  return __builtin_convertvector(v, v8bf);
}

// ---------------- kernel 1: fused QKV projection GEMM ----------------
// Y = X @ W^T + b ; X:[4096,1024] f32, W:[1024,1024] f32 -> bf16 [B,H,T,Dh]

#define BM 128
#define BN 64
#define BK 32
#define LDA 40   // 32 + 8 pad (80B row stride, 16B aligned)
#define LDB 40

__global__ __launch_bounds__(256)
void qkv_gemm_kernel(const float* __restrict__ X,
                     const float* __restrict__ Wq, const float* __restrict__ Bq,
                     const float* __restrict__ Wk, const float* __restrict__ Bk,
                     const float* __restrict__ Wv, const float* __restrict__ Bv,
                     __bf16* __restrict__ Qo, __bf16* __restrict__ Ko,
                     __bf16* __restrict__ Vo) {
  __shared__ __bf16 As[BM * LDA];
  __shared__ __bf16 Bs[BN * LDB];

  const int z = blockIdx.z;
  const float* W  = (z == 0) ? Wq : (z == 1) ? Wk : Wv;
  const float* Bb = (z == 0) ? Bq : (z == 1) ? Bk : Bv;
  __bf16* Dst     = (z == 0) ? Qo : (z == 1) ? Ko : Vo;

  const int m0 = blockIdx.x * BM;
  const int n0 = blockIdx.y * BN;
  const int t    = threadIdx.x;
  const int lane = t & 31;
  const int wave = t >> 5;
  const int wm = wave & 3;
  const int wn = wave >> 2;

  v8f acc[2][2] = {};

  const int ar = t >> 1;            // 0..127 : A row
  const int ah = (t & 1) * 16;      // A col half
  const int br = t >> 2;            // 0..63  : B row
  const int bq4 = (t & 3) * 8;      // B col quarter

  const float* srcA = X + (size_t)(m0 + ar) * D_MODEL + ah;
  const float* srcB = W + (size_t)(n0 + br) * D_MODEL + bq4;

  for (int kk = 0; kk < D_MODEL; kk += BK) {
    { // stage X f32 -> bf16 (packed cvt)
      union { v8f f; float4 q[2]; } u0, u1;
      u0.q[0] = ((const float4*)(srcA + kk))[0];
      u0.q[1] = ((const float4*)(srcA + kk))[1];
      u1.q[0] = ((const float4*)(srcA + kk))[2];
      u1.q[1] = ((const float4*)(srcA + kk))[3];
      __bf16* d = &As[ar * LDA + ah];
      *(v8bf*)(d)     = cvt8(u0.f);
      *(v8bf*)(d + 8) = cvt8(u1.f);
    }
    { // stage W rows (B[k][n] = W[n][k])
      union { v8f f; float4 q[2]; } u0;
      u0.q[0] = ((const float4*)(srcB + kk))[0];
      u0.q[1] = ((const float4*)(srcB + kk))[1];
      *(v8bf*)(&Bs[br * LDB + bq4]) = cvt8(u0.f);
    }
    if (kk + BK < D_MODEL) {  // hide next tile's HBM latency
      __builtin_prefetch(srcA + kk + BK, 0, 1);
      __builtin_prefetch(srcB + kk + BK, 0, 1);
    }
    __syncthreads();

    v16bf af[2], bfm[2];
#pragma unroll
    for (int i = 0; i < 2; ++i)
      af[i] = load_frag16(&As[(wm * 32 + i * 16) * LDA], LDA, lane);
#pragma unroll
    for (int j = 0; j < 2; ++j)
      bfm[j] = load_frag16(&Bs[(wn * 32 + j * 16) * LDB], LDB, lane);
#pragma unroll
    for (int i = 0; i < 2; ++i)
#pragma unroll
      for (int j = 0; j < 2; ++j)
        acc[i][j] = __builtin_amdgcn_wmma_f32_16x16x32_bf16(
            false, af[i], false, bfm[j], (short)0, acc[i][j], false, false);
    __syncthreads();
  }

  const int hf = (lane >> 4) & 1;
  const int nl = lane & 15;
#pragma unroll
  for (int i = 0; i < 2; ++i) {
#pragma unroll
    for (int j = 0; j < 2; ++j) {
      const int n = n0 + wn * 32 + j * 16 + nl;
      const int hh = n >> 6, dd = n & 63;
      v8bf ob = cvt8(acc[i][j] + Bb[n]);   // bias splat + packed convert
#pragma unroll
      for (int r = 0; r < 8; ++r) {
        const int m = m0 + wm * 32 + i * 16 + r + 8 * hf;
        const int bb = m >> 11, tt = m & (SEQ_T - 1);
        Dst[(((size_t)(bb * NHEAD + hh)) * SEQ_T + tt) * HEAD_DIM + dd] = ob[r];
      }
    }
  }
}

// ---------------- kernel 2: causal flash attention ----------------
// Q,K,V bf16 [B,H,T,64] -> O bf16 [B,T,1024]

#define KC  32
#define LDK 72   // 64 + 8 (TDM pad: 32 dwords -> pad 4 dwords)
#define LDV 40   // 32 + 8
#define LDP 40

__global__ __launch_bounds__(128)
void attn_kernel(const __bf16* __restrict__ Q, const __bf16* __restrict__ K,
                 const __bf16* __restrict__ V, __bf16* __restrict__ O) {
  __shared__ __bf16 Ks[KC * LDK];
  __shared__ __bf16 Vts[HEAD_DIM * LDV];    // V transposed: [d][k]
  __shared__ __bf16 Ps[4][16 * LDP];        // per-wave P tile

  const int qb = blockIdx.x * 64;
  const int h  = blockIdx.y;
  const int b  = blockIdx.z;
  const int t    = threadIdx.x;
  const int lane = t & 31;
  const int wave = t >> 5;
  const int hf = (lane >> 4) & 1;
  const int nl = lane & 15;

  const size_t headoff = ((size_t)(b * NHEAD + h)) * SEQ_T * HEAD_DIM;
  const __bf16* Qh = Q + headoff;
  const __bf16* Kh = K + headoff;
  const __bf16* Vh = V + headoff;

  // preload this wave's 16 Q rows as 2 A-fragments (d 0..31 / 32..63)
  v16bf qf[2];
  {
    const __bf16* qrow = Qh + (size_t)(qb + wave * 16 + nl) * HEAD_DIM + 8 * hf;
    union { v16bf v; v8bf q[2]; } u0, u1;
    u0.q[0] = *(const v8bf*)(qrow);
    u0.q[1] = *(const v8bf*)(qrow + 16);
    u1.q[0] = *(const v8bf*)(qrow + 32);
    u1.q[1] = *(const v8bf*)(qrow + 48);
    qf[0] = u0.v; qf[1] = u1.v;
  }

  const float scale = 0.125f;                 // 1/sqrt(64)
  const float LOG2E = 1.4426950408889634f;

  float mrow[8], lrow[8];
  v8f oacc[4] = {};
#pragma unroll
  for (int r = 0; r < 8; ++r) { mrow[r] = -3.0e38f; lrow[r] = 0.f; }

  const int ntile = qb / KC + 2;              // causal: K tiles up to qb+63

  const int sr = t >> 2;                      // 0..31 : K/V row
  const int sq = (t & 3) * 16;                // d quarter

  for (int kt = 0; kt < ntile; ++kt) {
    const int kbase = kt * KC;
    __syncthreads();
#if HAVE_TDM
    if (wave == 0) {   // TDM: DMA K tile [32][64] into padded LDS rows
      tdm_load_2d(Kh + (size_t)kbase * HEAD_DIM, lds_off(Ks),
                  /*tile_x=*/HEAD_DIM, /*tile_y=*/KC, /*stride_x=*/HEAD_DIM,
                  /*pad_ic: 32 dwords*/4, /*pad_ac: 4 dwords*/3);
    }
#else
    { // stage K tile [32][64] row-major
      const __bf16* src = Kh + (size_t)(kbase + sr) * HEAD_DIM + sq;
      __bf16* d = &Ks[sr * LDK + sq];
      *(v8bf*)(d)     = *(const v8bf*)(src);
      *(v8bf*)(d + 8) = *(const v8bf*)(src + 8);
    }
#endif
    { // stage V tile transposed -> Vts[d][k]
      const __bf16* src = Vh + (size_t)(kbase + sr) * HEAD_DIM + sq;
      v8bf a = *(const v8bf*)(src);
      v8bf c = *(const v8bf*)(src + 8);
#pragma unroll
      for (int i = 0; i < 8; ++i) {
        Vts[(sq + i) * LDV + sr]     = a[i];
        Vts[(sq + 8 + i) * LDV + sr] = c[i];
      }
    }
    if (kt + 1 < ntile) {
      __builtin_prefetch(Kh + (size_t)(kbase + KC + sr) * HEAD_DIM + sq, 0, 1);
      __builtin_prefetch(Vh + (size_t)(kbase + KC + sr) * HEAD_DIM + sq, 0, 1);
    }
#if HAVE_TDM
    if (wave == 0) __builtin_amdgcn_s_wait_tensorcnt(0);
#endif
    __syncthreads();

    // S = Q K^T  (two 16x16 tiles, K-dim = d in two 32-chunks)
    v8f s[2] = {};
#pragma unroll
    for (int nt = 0; nt < 2; ++nt) {
      v16bf k0 = load_frag16(&Ks[(nt * 16) * LDK], LDK, lane);
      v16bf k1 = load_frag16(&Ks[(nt * 16) * LDK] + 32, LDK, lane);
      s[nt] = __builtin_amdgcn_wmma_f32_16x16x32_bf16(
          false, qf[0], false, k0, (short)0, s[nt], false, false);
      s[nt] = __builtin_amdgcn_wmma_f32_16x16x32_bf16(
          false, qf[1], false, k1, (short)0, s[nt], false, false);
    }

    // scale + causal mask
    const int qrow0 = qb + wave * 16 + 8 * hf;
#pragma unroll
    for (int nt = 0; nt < 2; ++nt) {
      const int kcol = kbase + nt * 16 + nl;
#pragma unroll
      for (int r = 0; r < 8; ++r) {
        float v = s[nt][r] * scale;
        s[nt][r] = (kcol > qrow0 + r) ? -3.0e38f : v;
      }
    }

    // online softmax (row lives across 16 lanes of same half: xor 1,2,4,8)
#pragma unroll
    for (int r = 0; r < 8; ++r) {
      float mx = fmaxf(s[0][r], s[1][r]);
#pragma unroll
      for (int off = 8; off >= 1; off >>= 1)
        mx = fmaxf(mx, __shfl_xor(mx, off, 32));
      const float mnew = fmaxf(mrow[r], mx);
      const float alpha = exp2f((mrow[r] - mnew) * LOG2E);
      float p0 = exp2f((s[0][r] - mnew) * LOG2E);
      float p1 = exp2f((s[1][r] - mnew) * LOG2E);
      s[0][r] = p0; s[1][r] = p1;
      float rs = p0 + p1;
#pragma unroll
      for (int off = 8; off >= 1; off >>= 1)
        rs += __shfl_xor(rs, off, 32);
      lrow[r] = lrow[r] * alpha + rs;
      mrow[r] = mnew;
#pragma unroll
      for (int j = 0; j < 4; ++j)
        oacc[j][r] *= alpha;
    }

    // C-layout -> A-fragment re-layout via wave-private LDS (packed cvt)
    __bf16* pw = &Ps[wave][0];
    {
      v8bf pb0 = cvt8(s[0]);
      v8bf pb1 = cvt8(s[1]);
#pragma unroll
      for (int r = 0; r < 8; ++r) {
        pw[(r + 8 * hf) * LDP + nl]      = pb0[r];
        pw[(r + 8 * hf) * LDP + 16 + nl] = pb1[r];
      }
    }

    // O += P V
    v16bf pf = load_frag16(pw, LDP, lane);
#pragma unroll
    for (int j = 0; j < 4; ++j) {
      v16bf vf = load_frag16(&Vts[(j * 16) * LDV], LDV, lane);
      oacc[j] = __builtin_amdgcn_wmma_f32_16x16x32_bf16(
          false, pf, false, vf, (short)0, oacc[j], false, false);
    }
  }

  // normalize + store O bf16 in [B,T,C]
  v8f rinv;
#pragma unroll
  for (int r = 0; r < 8; ++r) rinv[r] = 1.f / lrow[r];
#pragma unroll
  for (int j = 0; j < 4; ++j) {
    const int dd = j * 16 + nl;
    v8bf ob = cvt8(oacc[j] * rinv);
#pragma unroll
    for (int r = 0; r < 8; ++r) {
      const int qrow = qb + wave * 16 + r + 8 * hf;
      O[((size_t)(b * SEQ_T + qrow)) * D_MODEL + h * HEAD_DIM + dd] = ob[r];
    }
  }
}

// ---------------- kernel 3: output projection GEMM ----------------
// Y = O @ Wo^T + bo ; O bf16 [4096,1024], Y f32.

__global__ __launch_bounds__(256)
void out_gemm_kernel(const __bf16* __restrict__ A, const float* __restrict__ W,
                     const float* __restrict__ Bias, float* __restrict__ Y) {
  __shared__ __bf16 As[BM * LDA];
  __shared__ __bf16 Bs[BN * LDB];

  const int m0 = blockIdx.x * BM;
  const int n0 = blockIdx.y * BN;
  const int t    = threadIdx.x;
  const int lane = t & 31;
  const int wave = t >> 5;
  const int wm = wave & 3;
  const int wn = wave >> 2;

  v8f acc[2][2] = {};

  const int ar = t >> 1;
  const int ah = (t & 1) * 16;
  const int br = t >> 2;
  const int bq4 = (t & 3) * 8;

  const __bf16* srcA = A + (size_t)(m0 + ar) * D_MODEL + ah;
  const float*  srcB = W + (size_t)(n0 + br) * D_MODEL + bq4;

  for (int kk = 0; kk < D_MODEL; kk += BK) {
#if HAVE_TDM
    if (wave == 0) {   // TDM: DMA A tile [128][32] bf16 into padded LDS rows
      tdm_load_2d(A + (size_t)m0 * D_MODEL + kk, lds_off(As),
                  /*tile_x=*/BK, /*tile_y=*/BM, /*stride_x=*/D_MODEL,
                  /*pad_ic: 16 dwords*/3, /*pad_ac: 4 dwords*/3);
    }
#else
    { // A already bf16: straight vector copy
      const __bf16* src = srcA + kk;
      __bf16* d = &As[ar * LDA + ah];
      *(v8bf*)(d)     = *(const v8bf*)(src);
      *(v8bf*)(d + 8) = *(const v8bf*)(src + 8);
    }
#endif
    { // Wo f32 -> bf16 (packed cvt)
      union { v8f f; float4 q[2]; } u0;
      u0.q[0] = ((const float4*)(srcB + kk))[0];
      u0.q[1] = ((const float4*)(srcB + kk))[1];
      *(v8bf*)(&Bs[br * LDB + bq4]) = cvt8(u0.f);
    }
    if (kk + BK < D_MODEL) {
      __builtin_prefetch(srcA + kk + BK, 0, 1);
      __builtin_prefetch(srcB + kk + BK, 0, 1);
    }
#if HAVE_TDM
    if (wave == 0) __builtin_amdgcn_s_wait_tensorcnt(0);
#endif
    __syncthreads();

    v16bf af[2], bfm[2];
#pragma unroll
    for (int i = 0; i < 2; ++i)
      af[i] = load_frag16(&As[(wm * 32 + i * 16) * LDA], LDA, lane);
#pragma unroll
    for (int j = 0; j < 2; ++j)
      bfm[j] = load_frag16(&Bs[(wn * 32 + j * 16) * LDB], LDB, lane);
#pragma unroll
    for (int i = 0; i < 2; ++i)
#pragma unroll
      for (int j = 0; j < 2; ++j)
        acc[i][j] = __builtin_amdgcn_wmma_f32_16x16x32_bf16(
            false, af[i], false, bfm[j], (short)0, acc[i][j], false, false);
    __syncthreads();
  }

  const int hf = (lane >> 4) & 1;
  const int nl = lane & 15;
#pragma unroll
  for (int i = 0; i < 2; ++i) {
#pragma unroll
    for (int j = 0; j < 2; ++j) {
      const int n = n0 + wn * 32 + j * 16 + nl;
      const float bias = Bias[n];
#pragma unroll
      for (int r = 0; r < 8; ++r) {
        const int m = m0 + wm * 32 + i * 16 + r + 8 * hf;
        Y[(size_t)m * D_MODEL + n] = acc[i][j][r] + bias;
      }
    }
  }
}

// ---------------- launch ----------------

extern "C" void kernel_launch(void* const* d_in, const int* in_sizes, int n_in,
                              void* d_out, int out_size, void* d_ws, size_t ws_size,
                              hipStream_t stream) {
  const float* X  = (const float*)d_in[0];
  const float* Wq = (const float*)d_in[1];
  const float* Bq = (const float*)d_in[2];
  const float* Wk = (const float*)d_in[3];
  const float* Bk = (const float*)d_in[4];
  const float* Wv = (const float*)d_in[5];
  const float* Bv = (const float*)d_in[6];
  const float* Wo = (const float*)d_in[7];
  const float* Bo = (const float*)d_in[8];
  float* Out = (float*)d_out;

  const size_t mat = (size_t)MTOT * D_MODEL;    // 4M elements
  __bf16* qws = (__bf16*)d_ws;
  __bf16* kws = qws + mat;
  __bf16* vws = kws + mat;
  __bf16* ows = vws + mat;

  dim3 g1(MTOT / BM, D_MODEL / BN, 3);
  qkv_gemm_kernel<<<g1, 256, 0, stream>>>(X, Wq, Bq, Wk, Bk, Wv, Bv,
                                          qws, kws, vws);

  dim3 g2(SEQ_T / 64, NHEAD, SEQ_B);
  attn_kernel<<<g2, 128, 0, stream>>>(qws, kws, vws, ows);

  dim3 g3(MTOT / BM, D_MODEL / BN, 1);
  out_gemm_kernel<<<g3, 256, 0, stream>>>(ows, Wo, Bo, Out);
}
